// LSTMCell_70815420776934
// MI455X (gfx1250) — matrix operand order repaired and checked by reading, not compiled, not run on hardware
//
#include <hip/hip_runtime.h>
#include <hip/hip_bf16.h>

typedef __attribute__((ext_vector_type(16))) __bf16 v16bf;
typedef __attribute__((ext_vector_type(8)))  float  v8f;

#define H_DIM       256
#define K_DIM       512
#define N4H        1024
#define ROWS_PER_WG  16
#define LDA         520   // padded K for LDS A tile (bf16 elements)

__device__ __forceinline__ float wave_sum32(float v) {
#pragma unroll
  for (int off = 16; off > 0; off >>= 1) v += __shfl_xor(v, off, 32);
  return v;
}
__device__ __forceinline__ float sigmoidf(float x) {
  return 1.0f / (1.0f + __expf(-x));
}

// ---------------------------------------------------------------------------
// Pack W[512,1024] f32 -> bf16 in this kernel's contiguous-k WMMA B-fragment
// layout: for col-tile ct (n/16), k-step kk (k/32):
//   lane = (n%16) + 16*(k%32 >= 16), half p = k%16
//   dst half index = (((ct*16 + kk)*32) + lane)*16 + p
// so each lane's fragment is 16 consecutive bf16 (32B, two b128 loads).
// ---------------------------------------------------------------------------
__global__ void pack_w_kernel(const float* __restrict__ W,
                              unsigned short* __restrict__ Wp) {
  int idx = blockIdx.x * blockDim.x + threadIdx.x;   // over 512*1024
  int k = idx >> 10;          // 0..511
  int n = idx & (N4H - 1);    // 0..1023
  int ct   = n >> 4;
  int kk   = k >> 5;
  int kin  = k & 31;
  int lane = (n & 15) + ((kin >= 16) ? 16 : 0);
  int p    = kin & 15;
  size_t dst = (((size_t)(ct * 16 + kk) * 32) + lane) * 16 + p;
  __bf16 bv = (__bf16)W[(size_t)k * N4H + n];
  Wp[dst] = __builtin_bit_cast(unsigned short, bv);
}

// ---------------------------------------------------------------------------
// Fused: concat(x,h)@W + bias -> block LN1 -> gates -> LN2 -> zoneout
// One WG = 16 rows x all 1024 columns. 8 waves, each wave owns 8 col-tiles.
// ---------------------------------------------------------------------------
__global__ void __launch_bounds__(256)
lstm_fused_kernel(const float* __restrict__ x,  const float* __restrict__ h,
                  const float* __restrict__ c,
                  const unsigned short* __restrict__ Wp,
                  const float* __restrict__ bias,
                  const float* __restrict__ g1, const float* __restrict__ b1,
                  const float* __restrict__ g2, const float* __restrict__ b2,
                  const float* __restrict__ mask_h,
                  const float* __restrict__ mask_c,
                  float* __restrict__ out_h, float* __restrict__ out_c) {
  __shared__ float accS[ROWS_PER_WG * N4H];                 // 64 KB
  __bf16* Ash = reinterpret_cast<__bf16*>(accS);            // aliased A tile

  const int tid  = threadIdx.x;
  const int wave = tid >> 5;
  const int lane = tid & 31;
  const int gr0  = blockIdx.x * ROWS_PER_WG;

  // ---- stage A = [x | h] rows gr0..gr0+15 into LDS as bf16 ----
  for (int idx = tid; idx < ROWS_PER_WG * K_DIM; idx += 256) {
    int row = idx >> 9;
    int col = idx & (K_DIM - 1);
    float v = (col < H_DIM) ? x[(size_t)(gr0 + row) * H_DIM + col]
                            : h[(size_t)(gr0 + row) * H_DIM + (col - H_DIM)];
    Ash[row * LDA + col] = (__bf16)v;
  }
  __syncthreads();

  // ---- GEMM: 16x1024 tile, wave owns col-tiles [wave*8, wave*8+8) ----
  v8f acc[8] = {};
  const int m          = lane & 15;
  const int kbase_lane = (lane < 16) ? 0 : 16;

  for (int kk = 0; kk < 16; ++kk) {            // K step of 32
    union { v16bf v; unsigned int u[8]; } a;
#pragma unroll
    for (int r = 0; r < 8; ++r) {
      int kx = (kk << 5) + kbase_lane + 2 * r;
      a.u[r] = *reinterpret_cast<const unsigned int*>(&Ash[m * LDA + kx]);
    }
#pragma unroll
    for (int t = 0; t < 8; ++t) {
      int ct = wave * 8 + t;
      union { v16bf v; uint4 q[2]; } b;
      const uint4* bp = reinterpret_cast<const uint4*>(
          Wp + ((((size_t)(ct * 16 + kk) * 32) + lane) << 4));
      b.q[0] = bp[0];
      b.q[1] = bp[1];
      acc[t] = __builtin_amdgcn_wmma_f32_16x16x32_bf16(
          false, a.v, false, b.v, (short)0, acc[t], false, false);
    }
  }
  __syncthreads();   // done with Ash; reuse the LDS as f32 accumulator store

  // ---- accumulators (+bias) -> LDS ----
  const int mrow = (lane >> 4) * 8;   // C layout: VGPR r -> row r (+8 hi lanes)
#pragma unroll
  for (int t = 0; t < 8; ++t) {
    int coln  = (wave * 8 + t) * 16 + (lane & 15);
    float badd = bias[coln];
#pragma unroll
    for (int r = 0; r < 8; ++r)
      accS[(r + mrow) * N4H + coln] = acc[t][r] + badd;
  }
  __syncthreads();

  // ---- epilogue: wave handles rows 2*wave, 2*wave+1 ----
#pragma unroll
  for (int rr = 0; rr < 2; ++rr) {
    const int row = wave * 2 + rr;
    const size_t grow = (size_t)(gr0 + row);
    const float* arow = &accS[row * N4H];

    // block LayerNorm 1: four independent 256-wide blocks
    float mean[4], rstd[4];
#pragma unroll
    for (int blk = 0; blk < 4; ++blk) {
      float s = 0.f, s2 = 0.f;
#pragma unroll
      for (int t = 0; t < 8; ++t) {
        float v = arow[blk * 256 + lane + 32 * t];
        s += v; s2 += v * v;
      }
      s  = wave_sum32(s);
      s2 = wave_sum32(s2);
      float mu  = s * (1.0f / 256.0f);
      float var = s2 * (1.0f / 256.0f) - mu * mu;
      mean[blk] = mu;
      rstd[blk] = rsqrtf(var + 1e-5f);
    }

    // gates + raw new_c, accumulate stats for LayerNorm 2
    float ncr[8], og[8];
    float s = 0.f, s2 = 0.f;
#pragma unroll
    for (int t = 0; t < 8; ++t) {
      int n = lane + 32 * t;
      float iv = (arow[      n] - mean[0]) * rstd[0] * g1[      n] + b1[      n];
      float jv = (arow[256 + n] - mean[1]) * rstd[1] * g1[256 + n] + b1[256 + n];
      float fv = (arow[512 + n] - mean[2]) * rstd[2] * g1[512 + n] + b1[512 + n];
      float ov = (arow[768 + n] - mean[3]) * rstd[3] * g1[768 + n] + b1[768 + n];
      float cold = c[grow * H_DIM + n];
      float v = cold * sigmoidf(fv + 1.0f) + sigmoidf(iv) * tanhf(jv);
      ncr[t] = v;
      og[t]  = ov;
      s += v; s2 += v * v;
    }
    s  = wave_sum32(s);
    s2 = wave_sum32(s2);
    float mu2   = s * (1.0f / 256.0f);
    float rstd2 = rsqrtf(s2 * (1.0f / 256.0f) - mu2 * mu2 + 1e-5f);

    // LN2, output gate, zoneout, store
#pragma unroll
    for (int t = 0; t < 8; ++t) {
      int n = lane + 32 * t;
      size_t gi = grow * H_DIM + n;
      float cn = (ncr[t] - mu2) * rstd2 * g2[n] + b2[n];
      float nh = tanhf(cn) * sigmoidf(og[t]);
      float oh = (mask_h[gi] < 0.7f) ? nh : h[gi];
      float oc = (mask_c[gi] < 0.7f) ? cn : c[gi];
      out_h[gi] = oh;
      out_c[gi] = oc;
    }
  }
}

extern "C" void kernel_launch(void* const* d_in, const int* in_sizes, int n_in,
                              void* d_out, int out_size, void* d_ws, size_t ws_size,
                              hipStream_t stream) {
  const float* x    = (const float*)d_in[0];
  const float* h    = (const float*)d_in[1];
  const float* c    = (const float*)d_in[2];
  const float* W    = (const float*)d_in[3];
  const float* bias = (const float*)d_in[4];
  const float* g1   = (const float*)d_in[5];
  const float* b1   = (const float*)d_in[6];
  const float* g2   = (const float*)d_in[7];
  const float* b2   = (const float*)d_in[8];
  const float* mh   = (const float*)d_in[9];
  const float* mc   = (const float*)d_in[10];

  const int B = in_sizes[1] / H_DIM;          // h is [B, 256]
  float* out_h = (float*)d_out;
  float* out_c = out_h + (size_t)B * H_DIM;   // tuple (new_h, new_c) concatenated
  unsigned short* Wp = (unsigned short*)d_ws; // 1 MB bf16 packed weights

  pack_w_kernel<<<(K_DIM * N4H) / 256, 256, 0, stream>>>(W, Wp);
  lstm_fused_kernel<<<B / ROWS_PER_WG, 256, 0, stream>>>(
      x, h, c, Wp, bias, g1, b1, g2, b2, mh, mc, out_h, out_c);
}
